// NCLMemory_26792005993047
// MI455X (gfx1250) — matrix operand order repaired and checked by reading, not compiled, not run on hardware
//
#include <hip/hip_runtime.h>
#include <hip/hip_bf16.h>
#include <cstddef>
#include <cstdint>

// ---------------- problem constants ----------------
#define B_      128
#define D_      512
#define K_      16384
#define MLA_    4096
#define LD_     (K_ + 1)      // 16385: [l_pos | l_neg]
#define KNN_    128
#define NNEG_   400
#define NHARD_  400
#define HITER_  5
#define HNGST_  1000
#define TINV_   20.0f          // 1/T
#define WPOS_   0.2f
#define WKNN_   0.00625f       // (1-0.2)/128

typedef __bf16 v16bf __attribute__((ext_vector_type(16)));
typedef __bf16 v8bf  __attribute__((ext_vector_type(8)));
typedef float  v8f   __attribute__((ext_vector_type(8)));
typedef unsigned int v4u  __attribute__((ext_vector_type(4)));
typedef int          v4i  __attribute__((ext_vector_type(4)));
typedef int          v8i  __attribute__((ext_vector_type(8)));

#if defined(__gfx1250__) && __has_builtin(__builtin_amdgcn_tensor_load_to_lds) && \
    __has_builtin(__builtin_amdgcn_s_wait_tensorcnt)
#define USE_TDM 1
#else
#define USE_TDM 0
#endif

// ---------------- helpers ----------------
__device__ __forceinline__ unsigned f2u(float x) {
    unsigned u = __float_as_uint(x);
    return u ^ ((u >> 31) ? 0xFFFFFFFFu : 0x80000000u);   // order-preserving
}
__device__ __forceinline__ float u2f(unsigned u) {
    unsigned v = u ^ ((u & 0x80000000u) ? 0x80000000u : 0xFFFFFFFFu);
    return __uint_as_float(v);
}

__device__ __forceinline__ float brsum(float v, float* sh) {
    int tid = threadIdx.x;
    sh[tid] = v; __syncthreads();
    for (int s = blockDim.x >> 1; s > 0; s >>= 1) {
        if (tid < s) sh[tid] += sh[tid + s];
        __syncthreads();
    }
    float r = sh[0]; __syncthreads(); return r;
}
__device__ __forceinline__ int brsumi(int v, int* sh) {
    int tid = threadIdx.x;
    sh[tid] = v; __syncthreads();
    for (int s = blockDim.x >> 1; s > 0; s >>= 1) {
        if (tid < s) sh[tid] += sh[tid + s];
        __syncthreads();
    }
    int r = sh[0]; __syncthreads(); return r;
}
__device__ __forceinline__ float brmax(float v, float* sh) {
    int tid = threadIdx.x;
    sh[tid] = v; __syncthreads();
    for (int s = blockDim.x >> 1; s > 0; s >>= 1) {
        if (tid < s) sh[tid] = fmaxf(sh[tid], sh[tid + s]);
        __syncthreads();
    }
    float r = sh[0]; __syncthreads(); return r;
}
__device__ __forceinline__ int brscan_excl(int v, int* sh, int* total) {
    int tid = threadIdx.x;
    sh[tid] = v; __syncthreads();
    for (int o = 1; o < (int)blockDim.x; o <<= 1) {
        int t = (tid >= o) ? sh[tid - o] : 0;
        __syncthreads();
        sh[tid] += t;
        __syncthreads();
    }
    int incl = sh[tid];
    *total = sh[blockDim.x - 1];
    __syncthreads();
    return incl - v;
}

// k-th LARGEST (1-based) over vals[0..n): 4-pass byte radix select via LDS histogram
__device__ unsigned radix_kth_desc(const float* __restrict__ vals, int n, int k,
                                   unsigned* hist, int* sh2) {
    int tid = threadIdx.x;
    unsigned prefix = 0;
    int krem = k;
    for (int pass = 0; pass < 4; ++pass) {
        int shift = 24 - pass * 8;
        for (int b = tid; b < 256; b += blockDim.x) hist[b] = 0u;
        __syncthreads();
        unsigned hmask = (pass == 0) ? 0u : (0xFFFFFFFFu << (shift + 8));
        for (int idx = tid; idx < n; idx += blockDim.x) {
            unsigned u = f2u(vals[idx]);
            if ((u & hmask) == prefix)
                atomicAdd(&hist[(u >> shift) & 255u], 1u);
        }
        __syncthreads();
        if (tid == 0) {
            int cum = 0; int b = 255;
            for (; b >= 0; --b) { cum += (int)hist[b]; if (cum >= krem) break; }
            if (b < 0) b = 0;
            sh2[0] = b;
            sh2[1] = krem - (cum - (int)hist[b]);
        }
        __syncthreads();
        prefix |= ((unsigned)sh2[0]) << shift;
        krem = sh2[1];
        __syncthreads();
    }
    return prefix;
}

__device__ __forceinline__ void perm_params(int i, int t, unsigned& off, unsigned& step) {
    unsigned h = (unsigned)(i * HITER_ + t) * 0x9E3779B9u + 0x85EBCA6Bu;
    h ^= h >> 16; h *= 0x7FEB352Du; h ^= h >> 15; h *= 0x846CA68Bu; h ^= h >> 16;
    off = h & (MLA_ - 1);
    unsigned h2 = h * 0x2545F491u + 0x9E3779B9u; h2 ^= h2 >> 16;
    step = (h2 & (MLA_ - 1)) | 1u;   // odd => modular permutation of [0,4096)
}

// ---------------- kernel 0: fp32 -> bf16 streaming conversion --------------------
__global__ __launch_bounds__(256) void k_cvt(const float* __restrict__ src,
                                             __bf16* __restrict__ dst, int n) {
    int base = (blockIdx.x * 256 + threadIdx.x) * 8;
    if (base >= n) return;
    const float4* p = (const float4*)(src + base);
    float4 a = p[0], b = p[1];
    v8bf o;
    o[0] = (__bf16)a.x; o[1] = (__bf16)a.y; o[2] = (__bf16)a.z; o[3] = (__bf16)a.w;
    o[4] = (__bf16)b.x; o[5] = (__bf16)b.y; o[6] = (__bf16)b.z; o[7] = (__bf16)b.w;
    *(v8bf*)(dst + base) = o;
}

// ---------------- kernel 1: l_pos = diag(q k^T), written to col 0 ----------------
__global__ __launch_bounds__(256) void k_lpos(const float* __restrict__ q,
                                              const float* __restrict__ k,
                                              float* __restrict__ inputs_raw) {
    __shared__ float shf[256];
    int i = blockIdx.x;
    float s = 0.f;
    for (int c = threadIdx.x; c < D_; c += blockDim.x)
        s += q[i * D_ + c] * k[i * D_ + c];
    float tot = brsum(s, shf);
    if (threadIdx.x == 0) inputs_raw[(size_t)i * LD_] = tot;
}

// ---------------- kernel 2: C[B][N] = Q[B][512] @ W[N][512]^T via bf16 WMMA ----------
// block = 256 thr (8 wave32); each block: 16 rows x 128 cols; wave w: 16x16 tile.
// Q tile staged into LDS by the Tensor Data Mover (TDM) where available.
__global__ __launch_bounds__(256) void k_gemm_qwT(const __bf16* __restrict__ Qbf,
                                                  const __bf16* __restrict__ Wbf,
                                                  float* __restrict__ C,
                                                  int ldc, int coff) {
    __shared__ __bf16 qs[16 * D_];                // 16 KB bf16 q tile, shared by 8 waves
    const int tid  = threadIdx.x;
    const int wave = tid >> 5;
    const int lane = tid & 31;
    const int lo   = lane & 15;
    const int hi   = lane >> 4;
    const int m0   = blockIdx.y * 16;
    const int n0   = blockIdx.x * 128 + wave * 16;

#if USE_TDM
    if (wave == 0) {
        // Tensor DMA descriptor (ISA 8.3/8.4): 2-D tile, 512 x 16 elems of 2 bytes.
        unsigned lds_off = (unsigned)(uintptr_t)(&qs[0]);
        unsigned long long ga = (unsigned long long)(uintptr_t)(Qbf + (size_t)m0 * D_);
        v4u g0 = { 1u,                                   // count=1, is_restore=0
                   lds_off,                              // lds_addr
                   (unsigned)(ga & 0xFFFFFFFFu),         // global_addr[31:0]
                   (unsigned)((ga >> 32) & 0x01FFFFFFu) | (2u << 30) }; // addr[56:32]|type=2
        v8i g1 = { (int)(1u << 16),          // workgroup_mask=0, data_size=1 (2B)
                   (int)(512u << 16),        // tensor_dim0[15:0] << 16 (dim0 = 512)
                   (int)(16u  << 16),        // tensor_dim0[31:16]=0 | tensor_dim1.lo=16
                   (int)(512u << 16),        // tensor_dim1.hi=0 | tile_dim0 = 512
                   16,                       // tile_dim1 = 16, tile_dim2 = 0
                   512,                      // tensor_dim0_stride.lo = 512
                   0, 0 };                   // stride hi / dim1_stride = 0
        v4i z4 = { 0, 0, 0, 0 };
#if __clang_major__ >= 23
        v8i z8 = { 0, 0, 0, 0, 0, 0, 0, 0 };
        __builtin_amdgcn_tensor_load_to_lds(g0, g1, z4, z4, z8, 0);
#else
        __builtin_amdgcn_tensor_load_to_lds(g0, g1, z4, z4, 0);
#endif
        __builtin_amdgcn_s_wait_tensorcnt(0);
    }
    // keep clang from assuming qs is uninitialized (TDM write is invisible to it)
    asm volatile("" :: "r"(&qs[0]) : "memory");
#else
    for (int idx = tid; idx < 16 * D_; idx += 256)
        qs[idx] = Qbf[(size_t)m0 * D_ + idx];
#endif
    __syncthreads();

    v8f acc0, acc1;
#pragma unroll
    for (int r = 0; r < 8; ++r) { acc0[r] = 0.0f; acc1[r] = 0.0f; }

    const __bf16* wrow = Wbf + (size_t)(n0 + lo) * D_;

#pragma unroll 2
    for (int k0 = 0; k0 < D_; k0 += 64) {
        __builtin_prefetch(wrow + ((k0 + 256) & (D_ - 1)), 0, 1);   // branch-free
#pragma unroll
        for (int u = 0; u < 2; ++u) {
            int kk = k0 + u * 32;
            // A frag (16x32 bf16): row = lo; elem e -> k = kk + 16*hi + e
            const v8bf* pa0 = (const v8bf*)&qs[lo * D_ + kk + hi * 16];
            const v8bf* pa1 = (const v8bf*)&qs[lo * D_ + kk + hi * 16 + 8];
            v16bf a = __builtin_shufflevector(pa0[0], pa1[0],
                                              0,1,2,3,4,5,6,7,8,9,10,11,12,13,14,15);
            // B frag (32x16 bf16): col = lo; elem e -> k = kk + 16*hi + e (32B contig)
            const v8bf* pb = (const v8bf*)(wrow + kk + hi * 16);
            v16bf b = __builtin_shufflevector(pb[0], pb[1],
                                              0,1,2,3,4,5,6,7,8,9,10,11,12,13,14,15);
            if (u == 0)
                acc0 = __builtin_amdgcn_wmma_f32_16x16x32_bf16(
                           false, a, false, b, (short)0, acc0, false, false);
            else
                acc1 = __builtin_amdgcn_wmma_f32_16x16x32_bf16(
                           false, a, false, b, (short)0, acc1, false, false);
        }
    }

    // C layout: VGPR r -> row m0 + r + 8*hi, col n0 + lo
#pragma unroll
    for (int r = 0; r < 8; ++r) {
        int row = m0 + r + 8 * hi;
        C[(size_t)row * ldc + coff + n0 + lo] = acc0[r] + acc1[r];
    }
}

// ---------------- kernel 3: row squared norms of memory / la_memory -------------
__global__ __launch_bounds__(256) void k_norms(const float* __restrict__ mem,
                                               const float* __restrict__ lam,
                                               float* __restrict__ mem_n2,
                                               float* __restrict__ la_n2) {
    __shared__ float shf[256];
    int r = blockIdx.x;
    const float* src = (r < K_) ? (mem + (size_t)r * D_) : (lam + (size_t)(r - K_) * D_);
    float s = 0.f;
    for (int c = threadIdx.x; c < D_; c += blockDim.x) { float v = src[c]; s += v * v; }
    float tot = brsum(s, shf);
    if (threadIdx.x == 0) {
        if (r < K_) mem_n2[r] = tot; else la_n2[r - K_] = tot;
    }
}

// ---------------- kernel 4: per-row top-128 stats (smooth-hot targets) -----------
__global__ __launch_bounds__(256) void k_topstats(const float* __restrict__ inputs_raw,
                                                  float* __restrict__ stats) {
    __shared__ unsigned hist[256];
    __shared__ int   sh2[2];
    __shared__ float shf[256];
    __shared__ int   shi[256];
    int i = blockIdx.x;
    const float* row = inputs_raw + (size_t)i * LD_;

    unsigned uk = radix_kth_desc(row, LD_, KNN_, hist, sh2);

    int   cnt = 0;
    float sum = 0.f, mx = -3.4e38f;
    for (int idx = threadIdx.x; idx < LD_; idx += blockDim.x) {
        float x = row[idx];
        unsigned u = f2u(x);
        mx = fmaxf(mx, x);
        if (u > uk) { cnt++; sum += x; }
    }
    int   cgt = brsumi(cnt, shi);
    float sgt = brsum(sum, shf);
    float rmx = brmax(mx, shf);
    if (threadIdx.x == 0) {
        float xk   = u2f(uk);
        int   ties = KNN_ - cgt;
        float stop = sgt + (float)ties * xk;
        unsigned u0 = f2u(row[0]);
        float c0 = (u0 > uk || (u0 == uk && ties > 0)) ? 1.f : 0.f;
        stats[i * 4 + 0] = stop;
        stats[i * 4 + 1] = c0;
        stats[i * 4 + 2] = rmx;
        stats[i * 4 + 3] = row[0];
    }
}

// ---------------- kernel 5: per-row bottom-400 negative indices (deterministic) ---
__global__ __launch_bounds__(256) void k_negsel(const float* __restrict__ inputs_raw,
                                                int* __restrict__ neg_idx) {
    __shared__ unsigned hist[256];
    __shared__ int sh2[2];
    __shared__ int shi[256];
    int i = blockIdx.x;
    const float* row = inputs_raw + (size_t)i * LD_ + 1;   // negatives only

    // 400th smallest == (K - 399)-th largest
    unsigned uthr = radix_kth_desc(row, K_, K_ - NNEG_ + 1, hist, sh2);

    const int chunk = K_ / 256;           // 64 per thread, ordered => deterministic
    const int start = threadIdx.x * chunk;

    int cnt_lt = 0;
    for (int j = 0; j < chunk; ++j)
        if (f2u(row[start + j]) < uthr) cnt_lt++;
    int total_lt;
    int off_lt = brscan_excl(cnt_lt, shi, &total_lt);
    int pos = off_lt;
    for (int j = 0; j < chunk; ++j)
        if (f2u(row[start + j]) < uthr) neg_idx[i * NNEG_ + (pos++)] = start + j;

    int cnt_eq = 0;
    for (int j = 0; j < chunk; ++j)
        if (f2u(row[start + j]) == uthr) cnt_eq++;
    int total_eq;
    int off_eq = brscan_excl(cnt_eq, shi, &total_eq);
    int loc = 0;
    for (int j = 0; j < chunk; ++j) {
        if (f2u(row[start + j]) == uthr) {
            int p = total_lt + off_eq + (loc++);
            if (p < NNEG_) neg_idx[i * NNEG_ + p] = start + j;
        }
    }
}

// ---------------- kernel 6: hard-negative similarities (cross-dot trick) ----------
__global__ __launch_bounds__(256) void k_cross(const float* __restrict__ mem,
                                               const float* __restrict__ lam,
                                               const float* __restrict__ inputs_raw,
                                               const float* __restrict__ la_sim,
                                               const float* __restrict__ mem_n2,
                                               const float* __restrict__ la_n2,
                                               const int* __restrict__ neg_idx,
                                               const int* __restrict__ epoch,
                                               float* __restrict__ hard_cand) {
    if (*epoch < HNGST_) return;
    int i = blockIdx.x / HITER_;
    int t = blockIdx.x % HITER_;
    unsigned off, step;
    perm_params(i, t, off, step);

    const int wv = threadIdx.x >> 5;
    const int ln = threadIdx.x & 31;

    for (int j = wv; j < NNEG_; j += 8) {
        int negid = neg_idx[i * NNEG_ + j];
        int sel   = (int)((off + (unsigned)j * step) & (MLA_ - 1));
        const float* nrow = mem + (size_t)negid * D_;
        const float* srow = lam + (size_t)sel * D_;
        float cr = 0.f;
#pragma unroll 4
        for (int v = 0; v < 16; ++v)
            cr += nrow[v * 32 + ln] * srow[v * 32 + ln];
#pragma unroll
        for (int o = 16; o > 0; o >>= 1)
            cr += __shfl_xor(cr, o, 32);
        if (ln == 0) {
            float nq = inputs_raw[(size_t)i * LD_ + 1 + negid];
            float sq = la_sim[(size_t)i * MLA_ + sel];
            float n2 = mem_n2[negid];
            float s2 = la_n2[sel];
            float sA = (nq + 2.f * sq)  * rsqrtf(n2       + 4.f * cr + 4.f * s2);
            float sB = (2.f * nq + sq)  * rsqrtf(4.f * n2 + 4.f * cr +       s2);
            size_t base = (size_t)i * (HITER_ * 2 * NNEG_) + (size_t)t * (2 * NNEG_) + 2 * j;
            hard_cand[base + 0] = sA;
            hard_cand[base + 1] = sB;
        }
    }
}

// ---------------- kernel 7: per-row LSE + loss ------------------------------------
__global__ __launch_bounds__(256) void k_final(const float* __restrict__ inputs_raw,
                                               const float* __restrict__ hard_cand,
                                               const float* __restrict__ stats,
                                               const int* __restrict__ epoch,
                                               float* __restrict__ loss) {
    __shared__ unsigned hist[256];
    __shared__ int   sh2[2];
    __shared__ float shf[256];
    __shared__ int   shi[256];
    int i = blockIdx.x;
    const bool useHard = (*epoch >= HNGST_);
    const float* row  = inputs_raw + (size_t)i * LD_;
    const float* hrow = hard_cand  + (size_t)i * (HITER_ * 2 * NNEG_);
    const int nh = HITER_ * 2 * NNEG_;   // 4000

    unsigned uh = 0; int cgtH = 0; float hmax = -3.4e38f;
    if (useHard) {
        uh = radix_kth_desc(hrow, nh, NHARD_, hist, sh2);
        int cnt = 0; float mx = -3.4e38f;
        for (int idx = threadIdx.x; idx < nh; idx += blockDim.x) {
            float x = hrow[idx];
            mx = fmaxf(mx, x);
            if (f2u(x) > uh) cnt++;
        }
        cgtH = brsumi(cnt, shi);
        hmax = brmax(mx, shf);
    }

    float mraw = stats[i * 4 + 2];
    if (useHard) mraw = fmaxf(mraw, hmax);
    float Ms = mraw * TINV_;

    float z = 0.f;
    for (int idx = threadIdx.x; idx < LD_; idx += blockDim.x)
        z += __expf(row[idx] * TINV_ - Ms);
    if (useHard) {
        for (int idx = threadIdx.x; idx < nh; idx += blockDim.x) {
            float x = hrow[idx];
            if (f2u(x) > uh) z += __expf(x * TINV_ - Ms);
        }
    }
    float Z = brsum(z, shf);

    if (threadIdx.x == 0) {
        if (useHard) {
            int ties = NHARD_ - cgtH;
            Z += (float)ties * __expf(u2f(uh) * TINV_ - Ms);
        }
        float lse   = __logf(Z) + Ms;
        float stop  = stats[i * 4 + 0] * TINV_;
        float c0    = stats[i * 4 + 1];
        float x0    = stats[i * 4 + 3] * TINV_;
        float sum_t = WPOS_ + WKNN_ * (128.f - c0);
        float dot_t = WPOS_ * x0 + WKNN_ * (stop - c0 * x0);
        loss[i] = sum_t * lse - dot_t;
    }
}

// ---------------- kernel 8: mean over rows (fixed-order, deterministic) -----------
__global__ __launch_bounds__(128) void k_mean(const float* __restrict__ loss,
                                              float* __restrict__ out) {
    __shared__ float shf[128];
    float v = loss[threadIdx.x];
    float s = brsum(v, shf);
    if (threadIdx.x == 0) out[0] = s / (float)B_;
}

// ---------------- launch ----------------------------------------------------------
extern "C" void kernel_launch(void* const* d_in, const int* in_sizes, int n_in,
                              void* d_out, int out_size, void* d_ws, size_t ws_size,
                              hipStream_t stream) {
    (void)in_sizes; (void)n_in; (void)out_size; (void)ws_size;
    const float* q     = (const float*)d_in[0];
    const float* kk    = (const float*)d_in[1];
    const float* mem   = (const float*)d_in[2];
    const float* lam   = (const float*)d_in[3];
    const int*   epoch = (const int*)d_in[4];
    float* out = (float*)d_out;

    // workspace layout (all region sizes are multiples of 4 floats -> 16B aligned)
    float* base       = (float*)d_ws;
    float* inputs_raw = base;                       size_t o = (size_t)B_ * LD_;
    float* la_sim     = base + o;                   o += (size_t)B_ * MLA_;
    float* mem_n2     = base + o;                   o += K_;
    float* la_n2      = base + o;                   o += MLA_;
    float* stats      = base + o;                   o += (size_t)B_ * 4;
    float* loss       = base + o;                   o += B_;
    float* hard_cand  = base + o;                   o += (size_t)B_ * HITER_ * 2 * NNEG_;
    int*   neg_idx    = (int*)(base + o);           o += (size_t)B_ * NNEG_;
    __bf16* qbf   = (__bf16*)(base + o);            o += (size_t)B_ * D_ / 2;
    __bf16* membf = (__bf16*)(base + o);            o += (size_t)K_ * D_ / 2;
    __bf16* labf  = (__bf16*)(base + o);

    // fp32 -> bf16 pre-conversion (one streaming pass; GEMMs then read half the bytes)
    k_cvt<<<(B_  * D_) / 2048, 256, 0, stream>>>(q,   qbf,   B_  * D_);
    k_cvt<<<(K_  * D_) / 2048, 256, 0, stream>>>(mem, membf, K_  * D_);
    k_cvt<<<(MLA_* D_) / 2048, 256, 0, stream>>>(lam, labf,  MLA_* D_);

    k_lpos<<<B_, 256, 0, stream>>>(q, kk, inputs_raw);
    k_gemm_qwT<<<dim3(K_ / 128, B_ / 16), 256, 0, stream>>>(qbf, membf, inputs_raw, LD_, 1);
    k_gemm_qwT<<<dim3(MLA_ / 128, B_ / 16), 256, 0, stream>>>(qbf, labf, la_sim, MLA_, 0);
    k_norms<<<K_ + MLA_, 256, 0, stream>>>(mem, lam, mem_n2, la_n2);
    k_topstats<<<B_, 256, 0, stream>>>(inputs_raw, stats);
    k_negsel<<<B_, 256, 0, stream>>>(inputs_raw, neg_idx);
    k_cross<<<B_ * HITER_, 256, 0, stream>>>(mem, lam, inputs_raw, la_sim,
                                             mem_n2, la_n2, neg_idx, epoch, hard_cand);
    k_final<<<B_, 256, 0, stream>>>(inputs_raw, hard_cand, stats, epoch, loss);
    k_mean<<<1, 128, 0, stream>>>(loss, out);
}
